// InteractionPPBlock_51694226374814
// MI455X (gfx1250) — matrix operand order repaired and checked by reading, not compiled, not run on hardware
//
#include <hip/hip_runtime.h>

// ---------------------------------------------------------------------------
// CDNA5 (gfx1250) wave32 WMMA implementation of the DimeNet++ interaction
// block.  Dense math: v_wmma_f32_16x16x32_f16.  K1 x-tile staged via the
// Tensor Data Mover (tensor_load_to_lds + s_wait_tensorcnt) when available.
// ---------------------------------------------------------------------------

typedef __attribute__((ext_vector_type(16))) _Float16 v16h;
typedef __attribute__((ext_vector_type(8)))  _Float16 v8h;
typedef __attribute__((ext_vector_type(8)))  float    v8f;
typedef __attribute__((ext_vector_type(4)))  unsigned int uint32x4;
typedef __attribute__((ext_vector_type(8)))  int      int32x8;
typedef __attribute__((ext_vector_type(4)))  int      int32x4;

#define WMMA_F16(acc, a, b) \
  __builtin_amdgcn_wmma_f32_16x16x32_f16(false, (a), false, (b), (short)0, (acc), false, false)

#if __has_builtin(__builtin_amdgcn_tensor_load_to_lds) && \
    __has_builtin(__builtin_amdgcn_s_wait_tensorcnt)
#define HAS_TDM 1
#else
#define HAS_TDM 0
#endif

// Fast swish: v * v_rcp_f32(1 + exp(-v))  (4 VALU ops vs full IEEE divide)
__device__ __forceinline__ float swishf(float v) {
  return v * __builtin_amdgcn_rcpf(1.0f + __expf(-v));
}

// A-fragment (16x32 f16): lane holds row m = lane&15.
// VGPR 0..7 hold K = k0 + (lane>>4)*16 + {0..15}.
__device__ __forceinline__ v16h load_a_frag(const _Float16* Atile, int astride,
                                            int m16, int k0, int lh) {
  const _Float16* p = Atile + m16 * astride + k0 + lh * 16;
  v8h lo = *(const v8h*)p;
  v8h hi = *(const v8h*)(p + 8);
  v16h a;
#pragma unroll
  for (int i = 0; i < 8; ++i) { a[i] = lo[i]; a[i + 8] = hi[i]; }
  return a;
}

// B-fragment (32x16 f16): lane holds column n = lane&15,
// K = k0 + (lane>>4)*16 + {0..15}.  W stored row-major (N, K) => contiguous.
__device__ __forceinline__ v16h load_b_frag(const _Float16* W, int wK,
                                            int n, int k0, int lh) {
  return *(const v16h*)(W + (size_t)n * wK + k0 + lh * 16);
}

// ---------------------------------------------------------------------------
// K0: zero the float accumulators (agg E*64 + hout N*192, contiguous in ws).
// ---------------------------------------------------------------------------
__global__ void __launch_bounds__(256) k_zero(float* __restrict__ p, size_t n) {
  size_t i = (size_t)blockIdx.x * blockDim.x + threadIdx.x;
  size_t stride = (size_t)gridDim.x * blockDim.x;
  for (; i < n; i += stride) p[i] = 0.0f;
}

// ---------------------------------------------------------------------------
// K1: per-edge front end.
//   t      = swish(x @ Wx^T + bx)            (E,384)
//   x_ji   = t[:, :192]                       -> ws
//   g      = rbf @ Wrbf2^T                    (E,192), folded in
//   x_kj64 = swish((t[:,192:] * g) @ Wdown^T) (E,64) f32 -> ws
// Block = 256 threads (8 waves), 64 edges.
// ---------------------------------------------------------------------------
__global__ void __launch_bounds__(256) k_edge_pre(
    const _Float16* __restrict__ x, const _Float16* __restrict__ rbf,
    const _Float16* __restrict__ Wrbf2, const _Float16* __restrict__ Wx,
    const _Float16* __restrict__ bx, const _Float16* __restrict__ Wdown,
    _Float16* __restrict__ xji, float* __restrict__ xkj64) {
  __shared__ __align__(32) _Float16 sx[64 * 208];
  __shared__ __align__(32) _Float16 st[64 * 208];
  __shared__ __align__(32) _Float16 srbf[64 * 8];
  __shared__ __align__(32) _Float16 swr[192 * 8];

  const int tid = threadIdx.x;
  const int e0  = blockIdx.x * 64;

#if HAS_TDM
  // TDM: DMA the 64x192 f16 x-tile into LDS with 208-element row pitch
  // (iterating descriptor: 64 iterations x one 192-element row).
  if (tid < 32) {
    const unsigned long long ga = (unsigned long long)(uintptr_t)(x + (size_t)e0 * 192);
    const unsigned int       la = (unsigned int)(uintptr_t)(void*)sx;
    uint32x4 g0;
    g0[0] = 1u;                                   // count=1 (valid user D#)
    g0[1] = la;                                   // lds_addr
    g0[2] = (unsigned int)ga;                     // global_addr[31:0]
    g0[3] = (unsigned int)((ga >> 32) & 0x1FFFFFFu) | (2u << 30);  // [56:32], type=2
    int32x8 g1;
    g1[0] = (1 << 16) | (1 << 19);                // data_size=2B, iterate_enable=1
    g1[1] = (192 & 0xFFFF) << 16;                 // tensor_dim0 = 192 (bits 79:48)
    g1[2] = (1 << 16);                            // tensor_dim1 = 1   (bits 111:80)
    g1[3] = (192 << 16);                          // tile_dim0 = 192   (bits 127:112)
    g1[4] = 1;                                    // tile_dim1 = 1, tile_dim2 = 0
    g1[5] = 192;                                  // tensor_dim0_stride lo32
    g1[6] = 0;
    g1[7] = 0;
    int32x4 g2;
    g2[0] = 1;                                    // tensor_dim2 (unused, iterate mode)
    g2[1] = 208;                                  // lds_addr_increment (elements)
    g2[2] = 192;                                  // global_addr_increment (elements)
    g2[3] = (63 << 16);                           // iterate_count = 63 -> 64 rows
    int32x4 g3 = {0, 0, 0, 0};
#if __clang_major__ >= 23
    int32x8 g4 = {0, 0, 0, 0, 0, 0, 0, 0};
    __builtin_amdgcn_tensor_load_to_lds(g0, g1, g2, g3, g4, 0);
#else
    __builtin_amdgcn_tensor_load_to_lds(g0, g1, g2, g3, 0);
#endif
  }
#endif

  // One-time L2 prefetch of the big weight matrix (first blocks warm it).
  for (int idx = tid; idx < 576; idx += 256)
    __builtin_prefetch(Wx + idx * 128, 0, 1);

  for (int idx = tid; idx < 64 * 8; idx += 256) srbf[idx] = rbf[(size_t)e0 * 8 + idx];
  for (int idx = tid; idx < 192 * 8; idx += 256) swr[idx] = Wrbf2[idx];

#if HAS_TDM
  if (tid < 32) __builtin_amdgcn_s_wait_tensorcnt(0);
#else
  for (int idx = tid; idx < 64 * 24; idx += 256) {
    int r = idx / 24, c = (idx % 24) * 8;
    *(v8h*)&sx[r * 208 + c] = *(const v8h*)&x[(size_t)(e0 + r) * 192 + c];
  }
#endif
  __syncthreads();

  const int w   = __builtin_amdgcn_readfirstlane(tid >> 5);  // scalar wave id
  const int lane = tid & 31, lh = lane >> 4, m16 = lane & 15;
  const int mt = w & 3;
  const int nbase = (w >> 2) * 12;  // waves 0-3 -> x_ji cols, waves 4-7 -> x_kj cols

  for (int j = 0; j < 12; ++j) {
    const int nc = (nbase + j) * 16;       // scalar
    const int n  = nc + m16;               // 0..383
    v8f acc = {};
#pragma unroll
    for (int k0 = 0; k0 < 192; k0 += 32) {
      v16h av = load_a_frag(sx + mt * 16 * 208, 208, m16, k0, lh);
      v16h bv = load_b_frag(Wx, 192, n, k0, lh);
      acc = WMMA_F16(acc, av, bv);
    }
    const float bn = (float)bx[n];
    if (nc < 192) {  // scalar branch
#pragma unroll
      for (int v = 0; v < 8; ++v) {
        const int r = mt * 16 + v + 8 * lh;
        xji[(size_t)(e0 + r) * 192 + n] = (_Float16)swishf(acc[v] + bn);
      }
    } else {
      const int nk = n - 192;
#pragma unroll
      for (int v = 0; v < 8; ++v) {
        const int r = mt * 16 + v + 8 * lh;
        float g = 0.0f;
#pragma unroll
        for (int k = 0; k < 8; ++k)
          g += (float)srbf[r * 8 + k] * (float)swr[nk * 8 + k];
        st[r * 208 + nk] = (_Float16)(swishf(acc[v] + bn) * g);
      }
    }
  }
  __syncthreads();

  // (64x192) @ Wdown^T (192->64), swish, f32 out
  const int nb2 = (w >> 2) * 2;
  for (int j = 0; j < 2; ++j) {
    const int n = (nb2 + j) * 16 + m16;  // 0..63
    v8f acc = {};
#pragma unroll
    for (int k0 = 0; k0 < 192; k0 += 32) {
      v16h av = load_a_frag(st + mt * 16 * 208, 208, m16, k0, lh);
      v16h bv = load_b_frag(Wdown, 192, n, k0, lh);
      acc = WMMA_F16(acc, av, bv);
    }
#pragma unroll
    for (int v = 0; v < 8; ++v) {
      const int r = mt * 16 + v + 8 * lh;
      xkj64[(size_t)(e0 + r) * 64 + n] = swishf(acc[v]);
    }
  }
}

// ---------------------------------------------------------------------------
// K2: triplet aggregation.
//   cos(theta) = (pji . pkj) / sqrt((pji.pkj)^2 + |pji x pkj|^2)  (no atan2/cos)
//   sbf  = legendre(cos)*norm (7) outer sbf_rad[e1] (7,6) -> 42, K-padded to 64
//   proj = sbf @ lin_sbf (42,64)  via v_wmma_f32_16x16x32_f16
//   agg[e3] += x_kj64[e1] * proj  (f32 atomics; agg = 64MB, L2-resident)
// Block = 256 threads (8 waves), 16 triplets per wave.
// ---------------------------------------------------------------------------
__global__ void __launch_bounds__(256) k_triplet(
    const float* __restrict__ pos_diff, const float* __restrict__ sbf_rad,
    const int* __restrict__ e1, const int* __restrict__ e2,
    const int* __restrict__ e3, const float* __restrict__ lin_sbf,
    const float* __restrict__ xkj64, float* __restrict__ agg) {
  __shared__ __align__(32) _Float16 slin[64 * 64];   // lin_sbf^T, K-padded
  __shared__ __align__(32) _Float16 sA[8 * 16 * 64]; // per-wave sbf tiles
  __shared__ int se1[8 * 16];
  __shared__ int se3[8 * 16];

  const int tid = threadIdx.x;
  for (int idx = tid; idx < 64 * 64; idx += 256) {
    const int nn = idx >> 6, kk = idx & 63;
    slin[nn * 64 + kk] = (kk < 42) ? (_Float16)lin_sbf[kk * 64 + nn] : (_Float16)0.0f;
  }

  const int w = __builtin_amdgcn_readfirstlane(tid >> 5);
  const int lane = tid & 31, lh = lane >> 4, m16 = lane & 15;
  const int t0 = (blockIdx.x * 8 + w) * 16;

  if (lane < 16) {
    const int ti = t0 + m16;
    const int i1 = e1[ti], i2 = e2[ti], i3 = e3[ti];
    se1[w * 16 + m16] = i1;
    se3[w * 16 + m16] = i3;
    const float pkx = pos_diff[(size_t)i1 * 3 + 0], pky = pos_diff[(size_t)i1 * 3 + 1],
                pkz = pos_diff[(size_t)i1 * 3 + 2];
    const float pjx = pos_diff[(size_t)i2 * 3 + 0], pjy = pos_diff[(size_t)i2 * 3 + 1],
                pjz = pos_diff[(size_t)i2 * 3 + 2];
    const float a  = pjx * pkx + pjy * pky + pjz * pkz;
    const float cx = pjy * pkz - pjz * pky;
    const float cy = pjz * pkx - pjx * pkz;
    const float cz = pjx * pky - pjy * pkx;
    const float b2 = cx * cx + cy * cy + cz * cz;
    const float ct = a * rsqrtf(fmaxf(a * a + b2, 1e-30f));

    float P[7];
    P[0] = 1.0f; P[1] = ct;
#pragma unroll
    for (int l = 2; l < 7; ++l)
      P[l] = ((2.0f * l - 1.0f) * ct * P[l - 1] - (float)(l - 1) * P[l - 2]) * (1.0f / (float)l);
    const float nrm[7] = {0.28209479f, 0.48860251f, 0.63078313f, 0.74635267f,
                          0.84628438f, 0.93560258f, 1.01710724f};

    _Float16* arow = sA + (w * 16 + m16) * 64;
#pragma unroll
    for (int s7 = 0; s7 < 7; ++s7) {
      const float pv = P[s7] * nrm[s7];
#pragma unroll
      for (int r6 = 0; r6 < 6; ++r6)
        arow[s7 * 6 + r6] = (_Float16)(pv * sbf_rad[(size_t)i1 * 42 + s7 * 6 + r6]);
    }
#pragma unroll
    for (int k = 42; k < 64; ++k) arow[k] = (_Float16)0.0f;
  }
  __syncthreads();

  const _Float16* Aw = sA + w * 16 * 64;
  for (int nt = 0; nt < 4; ++nt) {
    const int n = nt * 16 + m16;  // 0..63
    v8f acc = {};
#pragma unroll
    for (int k0 = 0; k0 < 64; k0 += 32) {
      v16h av = load_a_frag(Aw, 64, m16, k0, lh);
      v16h bv = load_b_frag(slin, 64, n, k0, lh);
      acc = WMMA_F16(acc, av, bv);
    }
#pragma unroll
    for (int v = 0; v < 8; ++v) {
      const int mrow = v + 8 * lh;
      const int i1 = se1[w * 16 + mrow];
      const int i3 = se3[w * 16 + mrow];
      const float msg = acc[v] * xkj64[(size_t)i1 * 64 + n];
      atomicAdd(&agg[(size_t)i3 * 64 + n], msg);
    }
  }
}

// ---------------------------------------------------------------------------
// K3: per-edge back end (up-proj, residual stacks, mid layer, outputs).
// ---------------------------------------------------------------------------
enum { MODE_STORE = 0, MODE_ADD = 1, MODE_SWISH_ADD_G = 2 };

// One dense stage for the whole block: 4 M-tiles x 12 N-tiles over 8 waves.
// Ends with a block barrier.  MODE is a template param -> no runtime branches.
template <int MODE>
__device__ __forceinline__ void dense_stage(
    const _Float16* __restrict__ A, int astride, int Klen,
    const _Float16* __restrict__ W, const _Float16* __restrict__ b,
    _Float16* __restrict__ dst, const _Float16* __restrict__ gadd, int tid) {
  const int w = __builtin_amdgcn_readfirstlane(tid >> 5);
  const int lane = tid & 31, lh = lane >> 4, m16 = lane & 15;
  const int mt = w & 3;
  const int nbase = (w >> 2) * 6;
  for (int j = 0; j < 6; ++j) {
    const int n = (nbase + j) * 16 + m16;  // 0..191
    v8f acc = {};
    for (int k0 = 0; k0 < Klen; k0 += 32) {
      v16h av = load_a_frag(A + mt * 16 * astride, astride, m16, k0, lh);
      v16h bv = load_b_frag(W, Klen, n, k0, lh);
      acc = WMMA_F16(acc, av, bv);
    }
    const float bn = b ? (float)b[n] : 0.0f;
#pragma unroll
    for (int v = 0; v < 8; ++v) {
      const int r = mt * 16 + v + 8 * lh;
      const float s = swishf(acc[v] + bn);
      if (MODE == MODE_STORE)
        dst[r * 208 + n] = (_Float16)s;
      else if (MODE == MODE_ADD)
        dst[r * 208 + n] = (_Float16)((float)dst[r * 208 + n] + s);
      else  // swish + global f16 row-major (stride 192) addend
        dst[r * 208 + n] = (_Float16)(s + (float)gadd[r * 192 + n]);
    }
  }
  __syncthreads();
}

__global__ void __launch_bounds__(256) k_edge_post(
    const float* __restrict__ agg, const _Float16* __restrict__ xji,
    const _Float16* __restrict__ x, const _Float16* __restrict__ Wup,
    const _Float16* __restrict__ Wlin, const _Float16* __restrict__ blin,
    const _Float16* __restrict__ rbw1, const _Float16* __restrict__ rbb1,
    const _Float16* __restrict__ rbw2, const _Float16* __restrict__ rbb2,
    const _Float16* __restrict__ raw1, const _Float16* __restrict__ rab1,
    const _Float16* __restrict__ raw2, const _Float16* __restrict__ rab2,
    const _Float16* __restrict__ rbf_out, const int* __restrict__ iidx,
    _Float16* __restrict__ outh, float* __restrict__ hout) {
  __shared__ __align__(32) _Float16 sh[64 * 208];
  __shared__ __align__(32) _Float16 sg[64 * 208];
  __shared__ __align__(32) _Float16 sa[64 * 64];

  const int tid = threadIdx.x;
  const int e0  = blockIdx.x * 64;

  for (int idx = tid; idx < 64 * 64; idx += 256)
    sa[idx] = (_Float16)agg[(size_t)e0 * 64 + idx];
  __syncthreads();

  _Float16* cur = sh;
  _Float16* alt = sg;

  // h = swish(agg @ Wup^T) + x_ji
  dense_stage<MODE_SWISH_ADD_G>(sa, 64, 64, Wup, nullptr, cur,
                                xji + (size_t)e0 * 192, tid);
  // residual_before (1 layer)
  dense_stage<MODE_STORE>(cur, 208, 192, rbw1, rbb1, alt, nullptr, tid);
  dense_stage<MODE_ADD>(alt, 208, 192, rbw2, rbb2, cur, nullptr, tid);
  // h = swish(h @ lin_w^T + lin_b) + x
  dense_stage<MODE_SWISH_ADD_G>(cur, 208, 192, Wlin, blin, alt,
                                x + (size_t)e0 * 192, tid);
  { _Float16* t = cur; cur = alt; alt = t; }
  // residual_after layer 0
  dense_stage<MODE_STORE>(cur, 208, 192, raw1, rab1, alt, nullptr, tid);
  dense_stage<MODE_ADD>(alt, 208, 192, raw2, rab2, cur, nullptr, tid);
  // residual_after layer 1
  dense_stage<MODE_STORE>(cur, 208, 192, raw1 + 192 * 192, rab1 + 192, alt, nullptr, tid);
  dense_stage<MODE_ADD>(alt, 208, 192, raw2 + 192 * 192, rab2 + 192, cur, nullptr, tid);

  // write h, accumulate node segment-sum (h * rbf_out) into f32 L2-resident buf
  for (int idx = tid; idx < 64 * 192; idx += 256) {
    const int r = idx / 192, c = idx % 192;
    const _Float16 hv = cur[r * 208 + c];
    outh[(size_t)(e0 + r) * 192 + c] = hv;
    const float p = (float)hv * (float)rbf_out[(size_t)(e0 + r) * 192 + c];
    atomicAdd(&hout[(size_t)iidx[e0 + r] * 192 + c], p);
  }
}

// ---------------------------------------------------------------------------
// K4: convert f32 node accumulator to f16 output region.
// ---------------------------------------------------------------------------
__global__ void __launch_bounds__(256) k_out2(const float* __restrict__ hout,
                                              _Float16* __restrict__ out2, size_t n) {
  size_t i = (size_t)blockIdx.x * blockDim.x + threadIdx.x;
  size_t stride = (size_t)gridDim.x * blockDim.x;
  for (; i < n; i += stride) out2[i] = (_Float16)hout[i];
}

// ---------------------------------------------------------------------------
extern "C" void kernel_launch(void* const* d_in, const int* in_sizes, int n_in,
                              void* d_out, int out_size, void* d_ws, size_t ws_size,
                              hipStream_t stream) {
  const _Float16* x        = (const _Float16*)d_in[0];
  const _Float16* rbf      = (const _Float16*)d_in[1];
  const float*    pos_diff = (const float*)d_in[2];
  const float*    sbf_rad  = (const float*)d_in[3];
  const int*      e1       = (const int*)d_in[5];
  const int*      e2       = (const int*)d_in[6];
  const int*      e3       = (const int*)d_in[7];
  const float*    lin_sbf  = (const float*)d_in[8];
  const _Float16* rbf_out  = (const _Float16*)d_in[9];
  const int*      iidx     = (const int*)d_in[10];
  const _Float16* Wrbf2    = (const _Float16*)d_in[12];
  const _Float16* Wx       = (const _Float16*)d_in[13];
  const _Float16* bx       = (const _Float16*)d_in[14];
  const _Float16* Wdown    = (const _Float16*)d_in[15];
  const _Float16* Wup      = (const _Float16*)d_in[16];
  const _Float16* Wlin     = (const _Float16*)d_in[17];
  const _Float16* blin     = (const _Float16*)d_in[18];
  const _Float16* rbw1     = (const _Float16*)d_in[19];
  const _Float16* rbb1     = (const _Float16*)d_in[20];
  const _Float16* rbw2     = (const _Float16*)d_in[21];
  const _Float16* rbb2     = (const _Float16*)d_in[22];
  const _Float16* raw1     = (const _Float16*)d_in[23];
  const _Float16* rab1     = (const _Float16*)d_in[24];
  const _Float16* raw2     = (const _Float16*)d_in[25];
  const _Float16* rab2     = (const _Float16*)d_in[26];

  const int E = in_sizes[0] / 192;
  const int T = in_sizes[5];
  const int N = 16384;  // num_nodes (device scalar; fixed by problem setup)

  // workspace layout: xji | xkj64 | agg | hout   (agg,hout contiguous for K0)
  char* ws = (char*)d_ws;
  _Float16* xji   = (_Float16*)ws;                       // E*192 f16
  float*    xkj64 = (float*)(ws + (size_t)E * 192 * 2);  // E*64 f32
  float*    agg   = xkj64 + (size_t)E * 64;              // E*64 f32
  float*    hout  = agg + (size_t)E * 64;                // N*192 f32

  _Float16* outh = (_Float16*)d_out;
  _Float16* out2 = outh + (size_t)E * 192;

  const size_t nzero = (size_t)E * 64 + (size_t)N * 192;
  k_zero<<<2048, 256, 0, stream>>>(agg, nzero);
  k_edge_pre<<<E / 64, 256, 0, stream>>>(x, rbf, Wrbf2, Wx, bx, Wdown, xji, xkj64);
  k_triplet<<<T / 128, 256, 0, stream>>>(pos_diff, sbf_rad, e1, e2, e3, lin_sbf,
                                         xkj64, agg);
  k_edge_post<<<E / 64, 256, 0, stream>>>(agg, xji, x, Wup, Wlin, blin,
                                          rbw1, rbb1, rbw2, rbb2,
                                          raw1, rab1, raw2, rab2,
                                          rbf_out, iidx, outh, hout);
  k_out2<<<4096, 256, 0, stream>>>(hout, out2, (size_t)N * 192);
}
